// MultiHeadAttention_16329465659530
// MI455X (gfx1250) — matrix operand order repaired and checked by reading, not compiled, not run on hardware
//
#include <hip/hip_runtime.h>

// ---------------- fixed problem shape ----------------
#define BB 2
#define SS 2048
#define EE 1024
#define HH 16
#define DK 64
#define MM (BB * SS)   // 4096 rows for all projection GEMMs

typedef __attribute__((ext_vector_type(16))) __bf16 v16bf;
typedef __attribute__((ext_vector_type(8)))  __bf16 v8bf;
typedef __attribute__((ext_vector_type(8)))  float  v8f;

// ---------------- WMMA helpers ----------------
// D = A(16x32 bf16) * B(32x16 bf16) + C(16x16 f32)
__device__ inline v8f wmma_bf16(v16bf a, v16bf b, v8f c) {
    return __builtin_amdgcn_wmma_f32_16x16x32_bf16(
        /*neg_a=*/false, a, /*neg_b=*/false, b,
        /*c_mod=*/(short)0, c, /*reuse_a=*/false, /*reuse_b=*/false);
}

// A-fragment (16x32, 16-bit): lane l holds row M=l%16, half=l/16,
// K run [k0+16*half, +16) contiguous (matches bfrag)
__device__ inline v16bf afrag(const __bf16* rowptr, int k0, int half) {
    v8bf lo = *(const v8bf*)(rowptr + k0 + 16 * half);
    v8bf hi = *(const v8bf*)(rowptr + k0 + 8 + 16 * half);
    v16bf a;
#pragma unroll
    for (int i = 0; i < 8; i++) { a[i] = lo[i]; a[i + 8] = hi[i]; }
    return a;
}

// B-fragment (32x16, 16-bit): lane l holds column N=l%16,
// K run [k0+16*half, +16) contiguous (same k order as afrag)
__device__ inline v16bf bfrag(const __bf16* rowptr, int k0, int half) {
    return *(const v16bf*)(rowptr + k0 + 16 * half);
}

// ---------------- CDNA5 async global->LDS copy (ASYNCcnt path) ----------------
// Each lane moves 16 bytes; LDS address = low 32 bits of the generic pointer
// (ISA: flat->LDS aperture mapping truncates to addr[31:0]).
__device__ inline void async_copy_b128(void* lds_dst, const void* gsrc) {
    unsigned lds = (unsigned)(uintptr_t)lds_dst;
    asm volatile("global_load_async_to_lds_b128 %0, %1, off"
                 :: "v"(lds), "v"(gsrc) : "memory");
}
__device__ inline void wait_async0() {
    asm volatile("s_wait_asynccnt 0x0" ::: "memory");
}

// ---------------- f32 -> bf16 convert ----------------
__global__ void __launch_bounds__(256) f2bf_kernel(const float* __restrict__ in,
                                                   __bf16* __restrict__ out, int n) {
    int i = blockIdx.x * 256 + threadIdx.x;
    if (i < n) out[i] = (__bf16)in[i];
}

// ---------------- GEMM: out[m,n] = sum_k A[m,k]*Bw[n,k] + bias[n] ----------------
// A: M x K bf16 row-major. Bw: N x K bf16 row-major (i.e. W, so result is X*W^T).
// layout 0/1 : write bf16 to (B,H,S,DK)      (Q / K projections)
// layout 2   : write bf16 to (B,H,DK,S)      (V projection, transposed per head)
// layout 3   : write f32 row-major M x N     (final output)
__global__ void __launch_bounds__(256) gemm_bias_wmma(
    const __bf16* __restrict__ A, const __bf16* __restrict__ Bw,
    const float* __restrict__ bias, __bf16* __restrict__ outb,
    float* __restrict__ outf, int M, int N, int K, int layout)
{
    const int lane = threadIdx.x & 31;
    const int wv   = threadIdx.x >> 5;   // 0..7
    const int half = lane >> 4;
    const int l16  = lane & 15;

    const int m0 = blockIdx.y * 128 + wv * 16;
    const int n0 = blockIdx.x * 64;

    const __bf16* arow  = A  + (size_t)(m0 + l16) * K;
    const __bf16* brow0 = Bw + (size_t)(n0 + l16) * K;

    v8f acc[4] = {};
    for (int k0 = 0; k0 < K; k0 += 32) {
        v16bf a = afrag(arow, k0, half);
#pragma unroll
        for (int nt = 0; nt < 4; nt++) {
            v16bf b = bfrag(brow0 + (size_t)nt * 16 * K, k0, half);
            acc[nt] = wmma_bf16(a, b, acc[nt]);
        }
    }

#pragma unroll
    for (int nt = 0; nt < 4; nt++) {
        const int col = n0 + nt * 16 + l16;
        const float bv = bias ? bias[col] : 0.0f;
#pragma unroll
        for (int r = 0; r < 8; r++) {
            const int row = m0 + r + 8 * half;     // C layout: VGPR r -> M=r (+8 for upper half)
            const float val = acc[nt][r] + bv;
            if (layout == 3) {
                outf[(size_t)row * N + col] = val;
            } else {
                const int bb = row >> 11;          // row = b*S + s, S=2048
                const int s  = row & (SS - 1);
                const int hh = col >> 6;           // col = h*64 + dd
                const int dd = col & (DK - 1);
                size_t idx;
                if (layout == 2)
                    idx = ((size_t)(bb * HH + hh) * DK + dd) * SS + s;   // (B,H,DK,S)
                else
                    idx = ((size_t)(bb * HH + hh) * SS + s) * DK + dd;   // (B,H,S,DK)
                outb[idx] = (__bf16)val;
            }
        }
    }
}

// ---------------- flash-style attention, double-buffered async LDS staging ----------------
// Q,K : (B,H,S,DK) bf16 ;  Vt : (B,H,DK,S) bf16 ;  Cc : (B,S,E) bf16 concat output
__global__ void __launch_bounds__(128) attn_wmma(
    const __bf16* __restrict__ Q, const __bf16* __restrict__ Kp,
    const __bf16* __restrict__ Vt, __bf16* __restrict__ Cc)
{
    __shared__ __align__(128) __bf16 kb[2][64 * 64];   // key block: [key][d]
    __shared__ __align__(128) __bf16 vt[2][64 * 64];   // V^T block: [d][key]
    __shared__ __align__(64)  __bf16 pt[4 * 16 * 64];  // P tiles per wave: [qrow][key]

    const int tid  = threadIdx.x;
    const int lane = tid & 31;
    const int wv   = tid >> 5;      // 0..3
    const int half = lane >> 4;
    const int l16  = lane & 15;

    const int bh = blockIdx.y;      // b*H + h
    const int b  = bh >> 4;
    const int h  = bh & (HH - 1);

    const __bf16* Qh = Q  + (size_t)bh * SS * DK;
    const __bf16* Kh = Kp + (size_t)bh * SS * DK;
    const __bf16* Vh = Vt + (size_t)bh * DK * SS;

    const int q0 = blockIdx.x * 64 + wv * 16;

    // cooperative-load mapping: 128 threads cover 64 rows x 2 half-rows (32 bf16 = 64B each)
    const int lr = tid >> 1;
    const int lh = tid & 1;

    auto issue_block = [&](int j, int buf) {
        const __bf16* gk = Kh + (size_t)(j * 64 + lr) * DK + lh * 32;
        const __bf16* gv = Vh + (size_t)lr * SS + j * 64 + lh * 32;
        __bf16* lk = &kb[buf][lr * 64 + lh * 32];
        __bf16* lv = &vt[buf][lr * 64 + lh * 32];
        async_copy_b128(lk,     gk);
        async_copy_b128(lk + 8, gk + 8);
        async_copy_b128(lv,     gv);
        async_copy_b128(lv + 8, gv + 8);
    };

    // Q fragments for this wave's 16 query rows (d = 0..31 and 32..63)
    const __bf16* qrow = Qh + (size_t)(q0 + l16) * DK;
    v16bf aq0 = afrag(qrow, 0,  half);
    v16bf aq1 = afrag(qrow, 32, half);

    v8f acc[4] = {};
    float mrow[8], lrow[8];
#pragma unroll
    for (int r = 0; r < 8; r++) { mrow[r] = -1e30f; lrow[r] = 0.0f; }

    issue_block(0, 0);   // prologue: prefetch first key block

    for (int j = 0; j < SS / 64; j++) {
        const int buf = j & 1;
        wait_async0();        // this thread's async copies (current buf) are in LDS
        __syncthreads();      // everyone's copies done AND previous iter's reads done

        if (j + 1 < SS / 64) issue_block(j + 1, buf ^ 1);   // overlap next staging

        const __bf16* kbuf = kb[buf];
        const __bf16* vbuf = vt[buf];

        // scores: S = Q * K^T  (contraction over d, contiguous in both)
        v8f sc[4];
#pragma unroll
        for (int nt = 0; nt < 4; nt++) {
            const __bf16* kr = kbuf + (nt * 16 + l16) * 64;
            v16bf b0 = *(const v16bf*)(kr + 16 * half);
            v16bf b1 = *(const v16bf*)(kr + 32 + 16 * half);
            v8f z = {};
            v8f t = wmma_bf16(aq0, b0, z);
            sc[nt] = wmma_bf16(aq1, b1, t);
#pragma unroll
            for (int r = 0; r < 8; r++) sc[nt][r] *= 0.125f;  // 1/sqrt(64)
        }

        // ---- online softmax (rows live at VGPR index across a 16-lane half) ----
        float mloc[8];
#pragma unroll
        for (int r = 0; r < 8; r++)
            mloc[r] = fmaxf(fmaxf(sc[0][r], sc[1][r]), fmaxf(sc[2][r], sc[3][r]));
#pragma unroll
        for (int mask = 1; mask <= 8; mask <<= 1)
#pragma unroll
            for (int r = 0; r < 8; r++)
                mloc[r] = fmaxf(mloc[r], __shfl_xor(mloc[r], mask, 32));

        float alpha[8];
#pragma unroll
        for (int r = 0; r < 8; r++) {
            float mn = fmaxf(mrow[r], mloc[r]);
            alpha[r] = __expf(mrow[r] - mn);
            mrow[r] = mn;
        }

        float srow[8];
#pragma unroll
        for (int r = 0; r < 8; r++) srow[r] = 0.0f;
#pragma unroll
        for (int nt = 0; nt < 4; nt++)
#pragma unroll
            for (int r = 0; r < 8; r++) {
                float p = __expf(sc[nt][r] - mrow[r]);
                sc[nt][r] = p;
                srow[r] += p;
            }
#pragma unroll
        for (int mask = 1; mask <= 8; mask <<= 1)
#pragma unroll
            for (int r = 0; r < 8; r++)
                srow[r] += __shfl_xor(srow[r], mask, 32);
#pragma unroll
        for (int r = 0; r < 8; r++) lrow[r] = alpha[r] * lrow[r] + srow[r];

        // rescale running output
#pragma unroll
        for (int dt = 0; dt < 4; dt++)
#pragma unroll
            for (int r = 0; r < 8; r++) acc[dt][r] *= alpha[r];

        // P: C-layout -> LDS -> A-fragment layout
        __bf16* pw = pt + wv * 16 * 64;
#pragma unroll
        for (int nt = 0; nt < 4; nt++)
#pragma unroll
            for (int r = 0; r < 8; r++)
                pw[(r + 8 * half) * 64 + nt * 16 + l16] = (__bf16)sc[nt][r];
        __syncthreads();

        v16bf ap0 = afrag(pw + l16 * 64, 0,  half);
        v16bf ap1 = afrag(pw + l16 * 64, 32, half);

        // O += P * V  (contraction over keys; Vt rows are d, contiguous in key)
#pragma unroll
        for (int dt = 0; dt < 4; dt++) {
            const __bf16* vr = vbuf + (dt * 16 + l16) * 64;
            v16bf b0 = *(const v16bf*)(vr + 16 * half);
            v16bf b1 = *(const v16bf*)(vr + 32 + 16 * half);
            acc[dt] = wmma_bf16(ap0, b0, acc[dt]);
            acc[dt] = wmma_bf16(ap1, b1, acc[dt]);
        }
    }

    // epilogue: normalize and write into concat (B,S,E) layout
#pragma unroll
    for (int dt = 0; dt < 4; dt++)
#pragma unroll
        for (int r = 0; r < 8; r++) {
            float o = acc[dt][r] / lrow[r];
            int row = q0 + r + 8 * half;
            int col = h * DK + dt * 16 + l16;
            Cc[(size_t)(b * SS + row) * EE + col] = (__bf16)o;
        }
}

// ---------------- host launcher ----------------
static inline char* carve(char*& p, size_t bytes) {
    char* r = p;
    p += (bytes + 255) & ~(size_t)255;
    return r;
}

extern "C" void kernel_launch(void* const* d_in, const int* in_sizes, int n_in,
                              void* d_out, int out_size, void* d_ws, size_t ws_size,
                              hipStream_t stream) {
    const float* q  = (const float*)d_in[0];
    const float* k  = (const float*)d_in[1];
    const float* v  = (const float*)d_in[2];
    const float* Wq = (const float*)d_in[3];
    const float* bq = (const float*)d_in[4];
    const float* Wk = (const float*)d_in[5];
    const float* bk = (const float*)d_in[6];
    const float* Wv = (const float*)d_in[7];
    const float* bv = (const float*)d_in[8];
    const float* Wo = (const float*)d_in[9];
    const float* bo = (const float*)d_in[10];

    const size_t NX = (size_t)MM * EE;   // 4096*1024
    const size_t NW = (size_t)EE * EE;   // 1024*1024

    char* p = (char*)d_ws;
    __bf16* qb  = (__bf16*)carve(p, NX * 2);
    __bf16* kb_ = (__bf16*)carve(p, NX * 2);
    __bf16* vb  = (__bf16*)carve(p, NX * 2);
    __bf16* wqb = (__bf16*)carve(p, NW * 2);
    __bf16* wkb = (__bf16*)carve(p, NW * 2);
    __bf16* wvb = (__bf16*)carve(p, NW * 2);
    __bf16* wob = (__bf16*)carve(p, NW * 2);
    __bf16* Qp  = (__bf16*)carve(p, NX * 2);   // (B,H,S,DK)
    __bf16* Kp  = (__bf16*)carve(p, NX * 2);   // (B,H,S,DK)
    __bf16* Vtp = (__bf16*)carve(p, NX * 2);   // (B,H,DK,S)
    __bf16* Cc  = (__bf16*)carve(p, NX * 2);   // concat (B,S,E)

    // f32 -> bf16 conversions
    f2bf_kernel<<<(int)((NX + 255) / 256), 256, 0, stream>>>(q, qb, (int)NX);
    f2bf_kernel<<<(int)((NX + 255) / 256), 256, 0, stream>>>(k, kb_, (int)NX);
    f2bf_kernel<<<(int)((NX + 255) / 256), 256, 0, stream>>>(v, vb, (int)NX);
    f2bf_kernel<<<(int)((NW + 255) / 256), 256, 0, stream>>>(Wq, wqb, (int)NW);
    f2bf_kernel<<<(int)((NW + 255) / 256), 256, 0, stream>>>(Wk, wkb, (int)NW);
    f2bf_kernel<<<(int)((NW + 255) / 256), 256, 0, stream>>>(Wv, wvb, (int)NW);
    f2bf_kernel<<<(int)((NW + 255) / 256), 256, 0, stream>>>(Wo, wob, (int)NW);

    // projections: X * W^T + b  with layout-specific stores
    dim3 gg(EE / 64, MM / 128);  // (16, 32)
    gemm_bias_wmma<<<gg, 256, 0, stream>>>(qb,  wqb, bq, Qp,  nullptr, MM, EE, EE, 0);
    gemm_bias_wmma<<<gg, 256, 0, stream>>>(kb_, wkb, bk, Kp,  nullptr, MM, EE, EE, 1);
    gemm_bias_wmma<<<gg, 256, 0, stream>>>(vb,  wvb, bv, Vtp, nullptr, MM, EE, EE, 2);

    // attention -> concat
    attn_wmma<<<dim3(SS / 64, BB * HH), 128, 0, stream>>>(Qp, Kp, Vtp, Cc);

    // output projection, f32 result straight to d_out
    gemm_bias_wmma<<<gg, 256, 0, stream>>>(Cc, wob, bo, nullptr, (float*)d_out,
                                           MM, EE, EE, 3);
}
